// NystromAttention_76493367542307
// MI455X (gfx1250) — compile-verified, matrix operand order checked by reading, not run on hardware
//
#include <hip/hip_runtime.h>
#include <hip/hip_bf16.h>

typedef __bf16 bf16;
typedef __attribute__((ext_vector_type(8)))  __bf16 v8bf;
typedef __attribute__((ext_vector_type(16))) __bf16 v16bf;
typedef __attribute__((ext_vector_type(8)))  float  v8f;
typedef __attribute__((__vector_size__(4 * sizeof(int)))) int v4i;

#define HD __device__ __forceinline__

#if defined(__has_builtin)
#if __has_builtin(__builtin_amdgcn_global_load_async_to_lds_b128)
#define USE_ASYNC_LDS 1
#endif
#endif
#ifndef USE_ASYNC_LDS
#define USE_ASYNC_LDS 0
#endif

constexpr int   Bb = 4, Nn = 8192, Cc = 512, Hh = 8, Dd = 64, Mm = 64;
constexpr float SCALE = 0.125f;  // D^-0.5

enum { MODE_F32 = 0, MODE_BF16 = 1, MODE_QKV = 2, MODE_PROJ = 3, MODE_ATTN = 4 };

HD v16bf make_frag(const bf16* p) {
    v8bf lo = *(const v8bf*)p;
    v8bf hi = *(const v8bf*)(p + 16);
    return __builtin_shufflevector(lo, hi, 0, 1, 2, 3, 4, 5, 6, 7,
                                   8, 9, 10, 11, 12, 13, 14, 15);
}

// ---------------------------------------------------------------------------
// Generic bf16 WMMA GEMM:  C = A(MxK, row-major, bf16) @ B(KxN, row-major, bf16)
// 128x64 workgroup tile, BK=64 (two WMMA K-steps per stage -> 8 v_wmma per
// barrier), 8 wave32 waves arranged 4x2, each owning a 32x32 accumulator block.
// A tile staged via GLOBAL_LOAD_ASYNC_TO_LDS_B128 when available (ASYNCcnt),
// B tile staged transposed via ds_store_b16 so its fragment mirrors A's.
// blockIdx.z selects the batch (per-(b,h) strides in elements).
// ---------------------------------------------------------------------------
__global__ __launch_bounds__(256)
void gemm_bf16_wmma(const bf16* __restrict__ A, const bf16* __restrict__ Bm,
                    void* __restrict__ Cv,
                    int M, int N, int K, int lda, int ldb, int ldc,
                    long long sA, long long sB, long long sC,
                    int mode, const float* __restrict__ bias,
                    bf16* __restrict__ qo, bf16* __restrict__ ko,
                    bf16* __restrict__ vo)
{
    constexpr int BM = 128, BN = 64, BK = 64, LDA = 72, LDB = 72;
    __shared__ bf16 sAt[BM * LDA];
    __shared__ bf16 sBt[BN * LDB];

    const int tid  = threadIdx.x;
    const int z    = blockIdx.z;
    A  += (size_t)z * (size_t)sA;
    Bm += (size_t)z * (size_t)sB;
    float* Cf = (float*)Cv + (size_t)z * (size_t)sC;
    bf16*  Cb = (bf16*)Cv  + (size_t)z * (size_t)sC;

    const int mBase = blockIdx.y * BM;
    const int nBase = blockIdx.x * BN;
    const int wave  = tid >> 5;
    const int lane  = tid & 31;
    const int wm    = wave & 3;           // 4 waves along M
    const int wn    = wave >> 2;          // 2 waves along N
    const int rowW  = wm * 32;
    const int colW  = wn * 32;
    const int fr    = lane & 15;          // fragment row/col within 16x16 tile
    const int kh    = (lane >> 4) * 8;    // K-half offset (also C-row offset)

    v8f acc[2][2] = {};

    for (int kb = 0; kb < K; kb += BK) {
        // --- stage A tile (BM x BK) into LDS, 16B chunks; OOB rows clamped
        //     (their accumulator rows are masked in the epilogue) ---
        for (int i = tid; i < BM * (BK / 8); i += 256) {
            int r  = i >> 3;              // BK/8 = 8 chunks per row
            int kc = (i & 7) << 3;
            int gm = mBase + r;
            if (gm >= M) gm = M - 1;      // branch-free clamp
            const bf16* gp = A + (size_t)gm * lda + kb + kc;
#if USE_ASYNC_LDS
            __builtin_amdgcn_global_load_async_to_lds_b128(
                (__attribute__((address_space(1))) v4i*)gp,
                (__attribute__((address_space(3))) v4i*)(&sAt[r * LDA + kc]),
                0, 0);
#else
            *(v8bf*)(&sAt[r * LDA + kc]) = *(const v8bf*)gp;
#endif
        }
        // --- stage B tile (BK x BN) transposed into LDS (column-contig K) ---
        for (int i = tid; i < BK * (BN / 8); i += 256) {
            int r  = i >> 3;
            int nc = (i & 7) << 3;
            v8bf val = *(const v8bf*)(Bm + (size_t)(kb + r) * ldb + nBase + nc);
#pragma unroll
            for (int e = 0; e < 8; e++) sBt[(nc + e) * LDB + r] = val[e];
            if (kb + BK < K)   // next B tile toward L2 (global_prefetch_b8)
                __builtin_prefetch(Bm + (size_t)(kb + BK + r) * ldb + nBase + nc,
                                   0, 1);
        }
#if USE_ASYNC_LDS
#if defined(__has_builtin) && __has_builtin(__builtin_amdgcn_s_wait_asynccnt)
        __builtin_amdgcn_s_wait_asynccnt(0);
#else
        asm volatile("s_wait_asynccnt 0x0" ::: "memory");
#endif
#endif
        __syncthreads();

#pragma unroll
        for (int s = 0; s < 2; s++) {     // two WMMA K-steps per stage
            const int ko_ = s * 32 + kh;
            v16bf af[2], bfr[2];
#pragma unroll
            for (int mt = 0; mt < 2; mt++)
                af[mt] = make_frag(&sAt[(rowW + mt * 16 + fr) * LDA + ko_]);
#pragma unroll
            for (int nt = 0; nt < 2; nt++)
                bfr[nt] = make_frag(&sBt[(colW + nt * 16 + fr) * LDB + ko_]);

#pragma unroll
            for (int mt = 0; mt < 2; mt++)
#pragma unroll
                for (int nt = 0; nt < 2; nt++)
                    acc[mt][nt] = __builtin_amdgcn_wmma_f32_16x16x32_bf16(
                        false, af[mt], false, bfr[nt], (short)0, acc[mt][nt],
                        false, false);
        }
        __syncthreads();
    }

    // --- epilogue (mode-dispatched, wave-uniform branch) ---
#pragma unroll
    for (int mt = 0; mt < 2; mt++) {
#pragma unroll
        for (int nt = 0; nt < 2; nt++) {
#pragma unroll
            for (int j = 0; j < 8; j++) {
                int m = mBase + rowW + mt * 16 + kh + j;
                int n = nBase + colW + nt * 16 + fr;
                if (m >= M) continue;
                float v = acc[mt][nt][j];
                if (mode == MODE_F32) {
                    Cf[(size_t)m * ldc + n] = v;
                } else if (mode == MODE_BF16) {
                    Cb[(size_t)m * ldc + n] = (bf16)v;
                } else if (mode == MODE_QKV) {
                    float val = v + bias[n];
                    int sel = n >> 9, rem = n & 511;
                    int head = rem >> 6, d = rem & 63;
                    if (sel == 0) val *= SCALE;   // fold D^-0.5 into q
                    int b = m >> 13, tok = m & (Nn - 1);
                    size_t off = ((size_t)(b * Hh + head) * Nn + tok) * Dd + d;
                    bf16* dst = (sel == 0) ? qo : ((sel == 1) ? ko : vo);
                    dst[off] = (bf16)val;
                } else if (mode == MODE_PROJ) {
                    Cf[(size_t)m * ldc + n] = v + bias[n];
                } else { // MODE_ATTN: (B,H)-batched 8192x64 -> (B,N,C) interleaved
                    int b = z >> 3, h = z & 7;
                    ((bf16*)Cv)[((size_t)b * Nn + m) * Cc + h * Dd + n] = (bf16)v;
                }
            }
        }
    }
}

// ---------------------------------------------------------------------------
// Small helper kernels
// ---------------------------------------------------------------------------
__global__ void cvt_f32_bf16(const float* __restrict__ in, bf16* __restrict__ out,
                             size_t n) {
    size_t i = (size_t)blockIdx.x * blockDim.x + threadIdx.x;
    if (i < n) out[i] = (bf16)in[i];
}

// kT[z][d*N+n] = k[z][n*D+d]    (z = b*H+h)
__global__ void transpose_k(const bf16* __restrict__ k, bf16* __restrict__ kT) {
    size_t i = (size_t)blockIdx.x * blockDim.x + threadIdx.x;
    if (i >= (size_t)Bb * Hh * Dd * Nn) return;
    int z = (int)(i >> 19);
    int rem = (int)(i & ((1u << 19) - 1));
    int d = rem >> 13, n = rem & (Nn - 1);
    kT[(size_t)z * 524288 + (size_t)d * Nn + n] =
        k[(size_t)z * 524288 + (size_t)n * Dd + d];
}

// Landmark gather: q_l (m x D) and k_l^T (D x m), idx = linspace(0,N-1,m)
__global__ void gather_lm(const bf16* __restrict__ q, const bf16* __restrict__ k,
                          bf16* __restrict__ ql, bf16* __restrict__ klT) {
    int z = blockIdx.x;
    for (int i = threadIdx.x; i < Mm * Dd; i += blockDim.x) {
        int j = i >> 6, d = i & 63;
        int idx = (int)((double)j * (double)(Nn - 1) / (double)(Mm - 1));
        size_t src = (size_t)z * 524288 + (size_t)idx * Dd + d;
        ql [(size_t)z * 4096 + j * 64 + d] = q[src];
        klT[(size_t)z * 4096 + d * 64 + j] = k[src];
    }
}

// Row softmax, 64-wide rows (kernel_1 / logits already scaled via q)
__global__ void softmax_rows64(const float* __restrict__ logits,
                               bf16* __restrict__ probs, int rows) {
    int r = blockIdx.x * blockDim.x + threadIdx.x;
    if (r >= rows) return;
    const float* p = logits + (size_t)r * 64;
    float mx = -3.0e38f;
#pragma unroll 8
    for (int i = 0; i < 64; i++) mx = fmaxf(mx, p[i]);
    float s = 0.f;
#pragma unroll 8
    for (int i = 0; i < 64; i++) s += __expf(p[i] - mx);
    float inv = 1.f / s;
    bf16* o = probs + (size_t)r * 64;
#pragma unroll 8
    for (int i = 0; i < 64; i++) o[i] = (bf16)(__expf(p[i] - mx) * inv);
}

// Row softmax, 8192-wide rows (kernel_3), one workgroup per row
__global__ __launch_bounds__(256)
void softmax_row8192(const float* __restrict__ logits, bf16* __restrict__ probs) {
    int r = blockIdx.x, tid = threadIdx.x;
    const float* p = logits + (size_t)r * Nn;
    __shared__ float red[256];
    float mx = -3.0e38f;
    for (int i = tid; i < Nn; i += 256) mx = fmaxf(mx, p[i]);
    red[tid] = mx; __syncthreads();
    for (int s = 128; s > 0; s >>= 1) {
        if (tid < s) red[tid] = fmaxf(red[tid], red[tid + s]);
        __syncthreads();
    }
    mx = red[0]; __syncthreads();
    float sum = 0.f;
    for (int i = tid; i < Nn; i += 256) sum += __expf(p[i] - mx);
    red[tid] = sum; __syncthreads();
    for (int s = 128; s > 0; s >>= 1) {
        if (tid < s) red[tid] += red[tid + s];
        __syncthreads();
    }
    float inv = 1.f / red[0]; __syncthreads();
    bf16* o = probs + (size_t)r * Nn;
    for (int i = tid; i < Nn; i += 256) o[i] = (bf16)(__expf(p[i] - mx) * inv);
}

// kernel_2 softmax + Newton-Schulz pseudo-inverse + out1 = pinv(k2) @ (k3 @ v)
// One workgroup per (b,h); all 64x64 matrices LDS-resident (4 x 16KB).
__global__ __launch_bounds__(256)
void k2_pinv_out1(const bf16* __restrict__ ql, const bf16* __restrict__ klT,
                  const float* __restrict__ k3v, bf16* __restrict__ out1) {
    int z = blockIdx.x, tid = threadIdx.x;
    __shared__ float S[4096], Z1[4096], Z2[4096], W[4096];

    for (int i = tid; i < 4096; i += 256) {
        Z1[i] = (float)ql [(size_t)z * 4096 + i];
        Z2[i] = (float)klT[(size_t)z * 4096 + i];
    }
    __syncthreads();
    // W = q_l @ k_l^T   (scale already folded into q)
    for (int i = tid; i < 4096; i += 256) {
        int r = i >> 6, c = i & 63;
        float s = 0.f;
        for (int k = 0; k < 64; k++) s = fmaf(Z1[r * 64 + k], Z2[k * 64 + c], s);
        W[i] = s;
    }
    __syncthreads();
    // S = row-softmax(W)
    if (tid < 64) {
        float mx = -3.0e38f;
        for (int j = 0; j < 64; j++) mx = fmaxf(mx, W[tid * 64 + j]);
        float sum = 0.f;
        for (int j = 0; j < 64; j++) sum += __expf(W[tid * 64 + j] - mx);
        float inv = 1.f / sum;
        for (int j = 0; j < 64; j++)
            S[tid * 64 + j] = __expf(W[tid * 64 + j] - mx) * inv;
    }
    __syncthreads();
    // Z0 = S^T / ||S||_1   (||S||_inf = 1 for a row-stochastic matrix)
    if (tid < 64) {
        float s = 0.f;
        for (int r = 0; r < 64; r++) s += S[r * 64 + tid];
        W[tid] = s;
    }
    __syncthreads();
    if (tid == 0) {
        float mx = 0.f;
        for (int c = 0; c < 64; c++) mx = fmaxf(mx, W[c]);
        W[64] = 1.f / mx;
    }
    __syncthreads();
    float alpha = W[64];
    __syncthreads();
    for (int i = tid; i < 4096; i += 256) {
        int r = i >> 6, c = i & 63;
        Z1[i] = alpha * S[c * 64 + r];
    }
    __syncthreads();
    // Newton-Schulz: Z <- Z (2I - S Z)
    float* Zc = Z1; float* Za = Z2;
    for (int it = 0; it < 14; it++) {
        for (int i = tid; i < 4096; i += 256) {
            int r = i >> 6, c = i & 63;
            float s = 0.f;
            for (int k = 0; k < 64; k++) s = fmaf(S[r * 64 + k], Zc[k * 64 + c], s);
            W[i] = ((r == c) ? 2.0f : 0.0f) - s;
        }
        __syncthreads();
        for (int i = tid; i < 4096; i += 256) {
            int r = i >> 6, c = i & 63;
            float s = 0.f;
            for (int k = 0; k < 64; k++) s = fmaf(Zc[r * 64 + k], W[k * 64 + c], s);
            Za[i] = s;
        }
        __syncthreads();
        float* t = Zc; Zc = Za; Za = t;
    }
    // out1 = Z @ (k3 @ v)
    for (int i = tid; i < 4096; i += 256) W[i] = k3v[(size_t)z * 4096 + i];
    __syncthreads();
    for (int i = tid; i < 4096; i += 256) {
        int r = i >> 6, c = i & 63;
        float s = 0.f;
        for (int k = 0; k < 64; k++) s = fmaf(Zc[r * 64 + k], W[k * 64 + c], s);
        out1[(size_t)z * 4096 + i] = (bf16)s;
    }
}

// ---------------------------------------------------------------------------
extern "C" void kernel_launch(void* const* d_in, const int* in_sizes, int n_in,
                              void* d_out, int out_size, void* d_ws, size_t ws_size,
                              hipStream_t stream) {
    const float* x      = (const float*)d_in[0];
    const float* qkv_w  = (const float*)d_in[1];
    const float* qkv_b  = (const float*)d_in[2];
    const float* proj_w = (const float*)d_in[3];
    const float* proj_b = (const float*)d_in[4];
    float* out = (float*)d_out;

    // -------- workspace carve-up (aligned 256B); sizes in bytes --------
    char* ws = (char*)d_ws;
    auto carve = [&](size_t bytes) {
        char* p = ws;
        ws += (bytes + 255) & ~(size_t)255;
        return p;
    };
    const size_t NTOK = (size_t)Bb * Nn;           // 32768
    const size_t QKVE = NTOK * Cc;                 // 16.8M elems (per q/k/v)
    bf16*  xbf   = (bf16*)carve(NTOK * Cc * 2);    // 32MB; reused as attn out
    bf16*  wqkv  = (bf16*)carve((size_t)Cc * 3 * Cc * 2);
    bf16*  wproj = (bf16*)carve((size_t)Cc * Cc * 2);
    bf16*  q     = (bf16*)carve(QKVE * 2);         // (B,H,N,D)
    bf16*  kbuf  = (bf16*)carve(QKVE * 2);         // k, later kernel_1 probs
    bf16*  v     = (bf16*)carve(QKVE * 2);
    bf16*  kT    = (bf16*)carve(QKVE * 2);         // (B,H,D,N), later k3 probs
    bf16*  ql    = (bf16*)carve((size_t)Bb * Hh * Mm * Dd * 2);
    bf16*  klT   = (bf16*)carve((size_t)Bb * Hh * Dd * Mm * 2);
    float* k3v   = (float*)carve((size_t)Bb * Hh * Mm * Dd * 4);
    bf16*  out1  = (bf16*)carve((size_t)Bb * Hh * Mm * Dd * 2);
    float* logit = (float*)carve(QKVE * 4);        // 64MB, reused k1->k3 logits
    bf16*  attn  = xbf;                            // reuse after QKV GEMM
    bf16*  k1p   = kbuf;                           // reuse after kT/gather
    bf16*  k3p   = kT;                             // reuse after kernel_3 logits

    // 1) fp32 -> bf16 conversions
    cvt_f32_bf16<<<(unsigned)((QKVE + 255) / 256), 256, 0, stream>>>(x, xbf, QKVE);
    cvt_f32_bf16<<<(unsigned)((Cc * 3 * Cc + 255) / 256), 256, 0, stream>>>(
        qkv_w, wqkv, (size_t)Cc * 3 * Cc);
    cvt_f32_bf16<<<(unsigned)((Cc * Cc + 255) / 256), 256, 0, stream>>>(
        proj_w, wproj, (size_t)Cc * Cc);

    // 2) QKV GEMM: (32768x512)@(512x1536)+b -> scatter bf16 q/k/v, q pre-scaled
    gemm_bf16_wmma<<<dim3(1536 / 64, 32768 / 128, 1), 256, 0, stream>>>(
        xbf, wqkv, nullptr, 32768, 1536, 512, 512, 1536, 0,
        0, 0, 0, MODE_QKV, qkv_b, q, kbuf, v);

    // 3) landmarks + k transpose
    gather_lm<<<Bb * Hh, 256, 0, stream>>>(q, kbuf, ql, klT);
    transpose_k<<<(unsigned)((QKVE + 255) / 256), 256, 0, stream>>>(kbuf, kT);

    // 4) kernel_1 logits: per (b,h) (8192x64)@(64x64) -> f32
    gemm_bf16_wmma<<<dim3(1, 8192 / 128, Bb * Hh), 256, 0, stream>>>(
        q, klT, logit, 8192, 64, 64, 64, 64, 64,
        524288, 4096, 524288, MODE_F32, nullptr, nullptr, nullptr, nullptr);
    softmax_rows64<<<(Bb * Hh * Nn) / 256, 256, 0, stream>>>(
        logit, k1p, Bb * Hh * Nn);

    // 5) kernel_3 logits: per (b,h) (64x64)@(64x8192) -> f32, then softmax
    gemm_bf16_wmma<<<dim3(8192 / 64, 1, Bb * Hh), 256, 0, stream>>>(
        ql, kT, logit, 64, 8192, 64, 64, 8192, 8192,
        4096, 524288, 524288, MODE_F32, nullptr, nullptr, nullptr, nullptr);
    softmax_row8192<<<Bb * Hh * Mm, 256, 0, stream>>>(logit, k3p);

    // 6) k3 @ v: per (b,h) (64x8192)@(8192x64) -> f32
    gemm_bf16_wmma<<<dim3(1, 1, Bb * Hh), 256, 0, stream>>>(
        k3p, v, k3v, 64, 64, 8192, 8192, 64, 64,
        524288, 524288, 4096, MODE_F32, nullptr, nullptr, nullptr, nullptr);

    // 7) kernel_2 softmax + iterative pinv + out1 = pinv @ (k3@v)
    k2_pinv_out1<<<Bb * Hh, 256, 0, stream>>>(ql, klT, k3v, out1);

    // 8) out = kernel_1 @ out1 -> (B,N,C)-interleaved bf16 attn buffer
    gemm_bf16_wmma<<<dim3(1, 8192 / 128, Bb * Hh), 256, 0, stream>>>(
        k1p, out1, attn, 8192, 64, 64, 64, 64, 0,
        524288, 4096, 0, MODE_ATTN, nullptr, nullptr, nullptr, nullptr);

    // 9) final projection: (32768x512)@(512x512)+b -> f32 d_out
    gemm_bf16_wmma<<<dim3(512 / 64, 32768 / 128, 1), 256, 0, stream>>>(
        attn, wproj, out, 32768, 512, 512, 512, 512, 512,
        0, 0, 0, MODE_PROJ, proj_b, nullptr, nullptr, nullptr);
}
